// QCGeMLinear_4312147165418
// MI455X (gfx1250) — compile-verified, matrix operand order checked by reading, not run on hardware
//
#include <hip/hip_runtime.h>

// ---------------------------------------------------------------------------
// W4A32 grouped-quant GEMM for MI455X (gfx1250, wave32, WMMA).
//   out[M,N] = x[M,K] * dequant(W_q)[K,N]
//   dequant: w = (nibble - zero[g,n]) * scale[g,n],  g = k / 128
// bf16 WMMA (16x16x32) with f32 accumulate; on-the-fly nibble dequant into an
// n-major bf16 LDS tile. Block tile 128x256, 8 waves, 64x64 per wave
// (4x4 WMMA accumulators -> 16 WMMA per 8 fragment loads per k-step).
// ---------------------------------------------------------------------------

typedef __attribute__((ext_vector_type(16))) __bf16 v16bf;
typedef __attribute__((ext_vector_type(8)))  float  v8f;

#define M_DIM 8192
#define K_DIM 4096
#define N_DIM 11008

#define BM 128
#define BN 256
#define BK 64   // quant group (128) constant within a K-tile

// float -> bf16, round-to-nearest-even
__device__ __forceinline__ unsigned short f2bf(float f) {
    unsigned int u = __float_as_uint(f);
    u += 0x7FFFu + ((u >> 16) & 1u);
    return (unsigned short)(u >> 16);
}

union FragAB { v16bf v; uint4 q[2]; };
union FragC  { v8f   v; float f[8]; };

__global__ __launch_bounds__(256)
void qgemm_w4_bf16_wmma(const float* __restrict__ x,
                        const int*   __restrict__ Wq,
                        const float* __restrict__ scales,
                        const float* __restrict__ zeros,
                        float*       __restrict__ out) {
    __shared__ unsigned short lds_a[BM][BK];   // bf16 A, row-major [m][k], 16 KB
    __shared__ unsigned short lds_b[BN][BK];   // bf16 B, n-major  [n][k], 32 KB

    const int tid  = threadIdx.x;
    const int lane = tid & 31;
    const int wave = tid >> 5;
    const int wm   = wave & 1;       // 2 wave-rows  -> 64 M-rows each
    const int wn   = wave >> 1;      // 4 wave-cols  -> 64 N-cols each
    const int m0   = blockIdx.y * BM;
    const int n0   = blockIdx.x * BN;

    v8f acc[4][4] = {};              // 4 M-tiles x 4 N-tiles of 16x16 f32

    // Each thread owns one B column of the block tile for the whole kernel.
    const int   ncolB = n0 + tid;            // tid in [0,256)
    const float* scol = scales + ncolB;
    const float* zcol = zeros  + ncolB;
    const int*   wcol = Wq     + ncolB;

    for (int k0 = 0; k0 < K_DIM; k0 += BK) {
        const int g   = k0 >> 7;     // quant group (constant over this tile)
        const int kp0 = k0 >> 3;     // packed-row base (8 nibbles / word)

        // ---- A: 128x64 fp32 -> bf16 LDS (8 x float4 per thread) ----
        #pragma unroll
        for (int it = 0; it < 8; ++it) {
            const int linear = it * 256 + tid;
            const int row    = linear >> 4;           // 16 float4 per row
            const int c4     = (linear & 15) << 2;
            const float4 v = *(const float4*)(x + (size_t)(m0 + row) * K_DIM + k0 + c4);
            const unsigned int p0 = (unsigned int)f2bf(v.x) | ((unsigned int)f2bf(v.y) << 16);
            const unsigned int p1 = (unsigned int)f2bf(v.z) | ((unsigned int)f2bf(v.w) << 16);
            *(uint2*)&lds_a[row][c4] = make_uint2(p0, p1);
        }

        // ---- B: 8x256 packed words -> 64x256 bf16 (n-major LDS) ----
        // Thread handles column `tid` for all 8 packed rows: scale/zero loaded once.
        const float s = scol[(size_t)g * N_DIM];
        const float z = zcol[(size_t)g * N_DIM];
        #pragma unroll
        for (int kp_l = 0; kp_l < 8; ++kp_l) {
            const int w = wcol[(size_t)(kp0 + kp_l) * N_DIM];
            unsigned int p[4];
            #pragma unroll
            for (int j = 0; j < 4; ++j) {
                const float f0 = ((float)((w >> (8 * j))     & 15) - z) * s;
                const float f1 = ((float)((w >> (8 * j + 4)) & 15) - z) * s;
                p[j] = (unsigned int)f2bf(f0) | ((unsigned int)f2bf(f1) << 16);
            }
            // 8 consecutive K-values of one N-column: one 128-bit LDS store
            *(uint4*)&lds_b[tid][kp_l << 3] = make_uint4(p[0], p[1], p[2], p[3]);
        }

        __syncthreads();

        // ---- compute: 2 K-steps of 32, 4x4 WMMA tiles per wave ----
        const int koff = (lane >> 4) << 3;   // half-wave K-chunk select (0 / 8)
        const int l16  = lane & 15;
        #pragma unroll
        for (int ks = 0; ks < BK; ks += 32) {
            FragAB a[4], b[4];
            #pragma unroll
            for (int i = 0; i < 4; ++i) {
                const unsigned short* p = &lds_a[wm * 64 + i * 16 + l16][ks + koff];
                a[i].q[0] = *(const uint4*)p;          // K  +0..7
                a[i].q[1] = *(const uint4*)(p + 16);   // K +16..23
            }
            #pragma unroll
            for (int j = 0; j < 4; ++j) {
                const unsigned short* p = &lds_b[wn * 64 + j * 16 + l16][ks + koff];
                b[j].q[0] = *(const uint4*)p;
                b[j].q[1] = *(const uint4*)(p + 16);
            }
            #pragma unroll
            for (int i = 0; i < 4; ++i)
                #pragma unroll
                for (int j = 0; j < 4; ++j)
                    acc[i][j] = __builtin_amdgcn_wmma_f32_16x16x32_bf16(
                        false, a[i].v, false, b[j].v,
                        (short)0, acc[i][j], false, false);
        }

        __syncthreads();
    }

    // ---- epilogue: C layout: VGPR r -> M = r + 8*(lane>=16), N = lane%16 ----
    const int mbase = m0 + wm * 64;
    const int nbase = n0 + wn * 64;
    const int moff  = (lane >> 4) << 3;
    const int ncol  = lane & 15;
    #pragma unroll
    for (int i = 0; i < 4; ++i) {
        #pragma unroll
        for (int j = 0; j < 4; ++j) {
            FragC c; c.v = acc[i][j];
            #pragma unroll
            for (int r = 0; r < 8; ++r) {
                out[(size_t)(mbase + i * 16 + moff + r) * N_DIM + nbase + j * 16 + ncol] = c.f[r];
            }
        }
    }
}

extern "C" void kernel_launch(void* const* d_in, const int* in_sizes, int n_in,
                              void* d_out, int out_size, void* d_ws, size_t ws_size,
                              hipStream_t stream) {
    const float* x      = (const float*)d_in[0];
    const int*   Wq     = (const int*)  d_in[1];
    const float* scales = (const float*)d_in[2];
    const float* zeros  = (const float*)d_in[3];
    float*       out    = (float*)d_out;

    dim3 grid(N_DIM / BN, M_DIM / BM);   // 43 x 64 blocks
    dim3 block(256);
    qgemm_w4_bf16_wmma<<<grid, block, 0, stream>>>(x, Wq, scales, zeros, out);

    (void)in_sizes; (void)n_in; (void)out_size; (void)d_ws; (void)ws_size;
}
